// SENDModel_18382460027458
// MI455X (gfx1250) — matrix operand order repaired and checked by reading, not compiled, not run on hardware
//
#include <hip/hip_runtime.h>

typedef _Float16 half_t;
typedef __attribute__((ext_vector_type(16))) _Float16 v16h;
typedef __attribute__((ext_vector_type(8)))  _Float16 v8h;
typedef __attribute__((ext_vector_type(8)))  float    v8f;

union V16 { v16h v; struct { v8h lo, hi; } p; };

__device__ __forceinline__ v8h cvt8(const float* __restrict__ p) {
  float4 f = *(const float4*)p;
  float4 g = *(const float4*)(p + 4);
  v8h r;
  r[0] = (_Float16)f.x; r[1] = (_Float16)f.y; r[2] = (_Float16)f.z; r[3] = (_Float16)f.w;
  r[4] = (_Float16)g.x; r[5] = (_Float16)g.y; r[6] = (_Float16)g.z; r[7] = (_Float16)g.w;
  return r;
}

struct Frag {
  V16 a[2];
  V16 b[4];
};

__device__ __forceinline__ void load_frags(Frag& f,
                                           const float* __restrict__ a0p,
                                           const float* __restrict__ a1p,
                                           const half_t* const wp[4],
                                           int k0, int hs) {
  int c0 = k0 + hs * 16;
  int c1 = k0 + hs * 16 + 8;
  f.a[0].p.lo = cvt8(a0p + c0);
  f.a[0].p.hi = cvt8(a0p + c1);
  f.a[1].p.lo = cvt8(a1p + c0);
  f.a[1].p.hi = cvt8(a1p + c1);
  int bc = k0 + hs * 16;
#pragma unroll
  for (int j = 0; j < 4; ++j) {
    f.b[j].p.lo = *(const v8h*)(wp[j] + bc);
    f.b[j].p.hi = *(const v8h*)(wp[j] + bc + 8);
  }
}

__device__ __forceinline__ void mma8(v8f (&acc)[2][4], const Frag& f) {
#pragma unroll
  for (int i = 0; i < 2; ++i)
#pragma unroll
    for (int j = 0; j < 4; ++j)
      acc[i][j] = __builtin_amdgcn_wmma_f32_16x16x32_f16(
          false, f.a[i].v, false, f.b[j].v, (short)0, acc[i][j], false, false);
}

// ---------------------------------------------------------------------------
// Generic WMMA GEMM: Out[M,N] = A[M,K](f32) @ W[N,K](f16)^T + bias (+res)(+relu)
// Wave tile 32(M) x 64(N): acc[2][4] 16x16 tiles, K stepped by 32.
// Main loop is branchless + double-buffered; K%32 handled in one guarded tail.
// ---------------------------------------------------------------------------
__global__ __launch_bounds__(256)
void gemm_ws_kernel(const float* __restrict__ A, int lda,
                    const half_t* __restrict__ W, int ldw,
                    const float* __restrict__ bias,
                    const float* __restrict__ res, int ldr,
                    float* __restrict__ Out, int ldo,
                    int M, int N, int K, int relu)
{
  int wave  = (int)(blockIdx.x) * 8 + (threadIdx.x >> 5);
  int lane  = threadIdx.x & 31;
  int tilesM = (M + 31) >> 5;
  int tilesN = (N + 63) >> 6;
  if (wave >= tilesM * tilesN) return;
  int tM = (wave % tilesM) << 5;
  int tN = (wave / tilesM) << 6;
  int row16 = lane & 15;
  int hs    = lane >> 4;

  int m0 = tM + row16;      if (m0 >= M) m0 = M - 1;
  int m1 = tM + 16 + row16; if (m1 >= M) m1 = M - 1;
  const float* a0p = A + (size_t)m0 * lda;
  const float* a1p = A + (size_t)m1 * lda;

  const half_t* wp[4];
#pragma unroll
  for (int j = 0; j < 4; ++j) {
    int n = tN + j * 16 + row16; if (n >= N) n = N - 1;
    wp[j] = W + (size_t)n * ldw;
  }

  v8f zf = {};
  v8f acc[2][4];
#pragma unroll
  for (int i = 0; i < 2; ++i)
#pragma unroll
    for (int j = 0; j < 4; ++j) acc[i][j] = zf;

  int kfull = K & ~31;      // whole 32-chunks (even count for all K used here)
  if (kfull >= 64) {
    Frag f0, f1;
    load_frags(f0, a0p, a1p, wp, 0, hs);
    int k0 = 0;
    for (; k0 + 64 < kfull; k0 += 64) {
      load_frags(f1, a0p, a1p, wp, k0 + 32, hs);
      mma8(acc, f0);
      load_frags(f0, a0p, a1p, wp, k0 + 64, hs);
      mma8(acc, f1);
    }
    load_frags(f1, a0p, a1p, wp, k0 + 32, hs);
    mma8(acc, f0);
    mma8(acc, f1);
  } else {
    Frag f0;
    for (int k0 = 0; k0 < kfull; k0 += 32) {
      load_frags(f0, a0p, a1p, wp, k0, hs);
      mma8(acc, f0);
    }
  }

  if (K & 31) {             // guarded tail (requires K % 8 == 0; true here)
    int k0 = kfull;
    int c0 = k0 + hs * 16;
    int c1 = k0 + hs * 16 + 8;
    v8h zh = {};
    Frag f;
    f.a[0].p.lo = (c0 < K) ? cvt8(a0p + c0) : zh;
    f.a[0].p.hi = (c1 < K) ? cvt8(a0p + c1) : zh;
    f.a[1].p.lo = (c0 < K) ? cvt8(a1p + c0) : zh;
    f.a[1].p.hi = (c1 < K) ? cvt8(a1p + c1) : zh;
    int bc = k0 + hs * 16;
    bool bok = bc < K;
#pragma unroll
    for (int j = 0; j < 4; ++j) {
      if (bok) {
        f.b[j].p.lo = *(const v8h*)(wp[j] + bc);
        f.b[j].p.hi = *(const v8h*)(wp[j] + bc + 8);
      } else {
        f.b[j].p.lo = zh; f.b[j].p.hi = zh;
      }
    }
    mma8(acc, f);
  }

#pragma unroll
  for (int j = 0; j < 4; ++j) {
    int n = tN + j * 16 + row16;
    if (n >= N) continue;
    float bv = bias ? bias[n] : 0.0f;
#pragma unroll
    for (int i = 0; i < 2; ++i) {
#pragma unroll
      for (int r = 0; r < 8; ++r) {
        int m = tM + i * 16 + r + 8 * hs;
        if (m >= M) continue;
        float v = acc[i][j][r] + bv;
        if (res)  v += res[(size_t)m * ldr + n];
        if (relu) v = fmaxf(v, 0.0f);
        Out[(size_t)m * ldo + n] = v;
      }
    }
  }
}

// ---------------------------------------------------------------------------
// Flash attention, one wave per (b, h, 16-row Q tile). Head dim 128, f16 WMMA.
// ---------------------------------------------------------------------------
__global__ __launch_bounds__(256)
void attn_kernel(const half_t* __restrict__ qf, const half_t* __restrict__ kf,
                 const half_t* __restrict__ vt, float* __restrict__ out,
                 int Bn, int H, int L)
{
  __shared__ half_t pscr[8][512];   // per-wave 16x32 f16 P tile
  int lane = threadIdx.x & 31;
  int wib  = threadIdx.x >> 5;
  int wave = (int)(blockIdx.x) * 8 + wib;
  int qtiles = L >> 4;
  if (wave >= Bn * H * qtiles) return;
  int qt = wave % qtiles;
  int bh = wave / qtiles;
  int h  = bh % H;
  int b  = bh / H;
  int row16 = lane & 15;
  int hs    = lane >> 4;

  const half_t* qbase = qf + (size_t)bh * L * 128;
  const half_t* kbase = kf + (size_t)bh * L * 128;
  const half_t* vbase = vt + (size_t)bh * 128 * L;

  // Preload Q A-fragments for the 4 head-dim K-steps
  V16 qa[4];
  {
    const half_t* qr = qbase + (size_t)(qt * 16 + row16) * 128;
#pragma unroll
    for (int kd = 0; kd < 4; ++kd) {
      qa[kd].p.lo = *(const v8h*)(qr + kd * 32 + hs * 16);
      qa[kd].p.hi = *(const v8h*)(qr + kd * 32 + hs * 16 + 8);
    }
  }

  v8f zf = {};
  v8f o[8];
#pragma unroll
  for (int j = 0; j < 8; ++j) o[j] = zf;
  float mx[8], sm[8];
#pragma unroll
  for (int i = 0; i < 8; ++i) { mx[i] = -3.0e38f; sm[i] = 0.0f; }

  const float scale = 0.08838834764831845f;  // 1/sqrt(128)
  int ktiles = L >> 5;
  for (int kt = 0; kt < ktiles; ++kt) {
    // ---- scores: two 16x16 tiles over 32 keys ----
    v8f s0 = zf, s1 = zf;
    const half_t* k0r = kbase + (size_t)(kt * 32 + row16) * 128;
    const half_t* k1r = k0r + 16 * 128;
#pragma unroll
    for (int kd = 0; kd < 4; ++kd) {
      V16 b0, b1;
      b0.p.lo = *(const v8h*)(k0r + kd * 32 + hs * 16);
      b0.p.hi = *(const v8h*)(k0r + kd * 32 + hs * 16 + 8);
      b1.p.lo = *(const v8h*)(k1r + kd * 32 + hs * 16);
      b1.p.hi = *(const v8h*)(k1r + kd * 32 + hs * 16 + 8);
      s0 = __builtin_amdgcn_wmma_f32_16x16x32_f16(false, qa[kd].v, false, b0.v, (short)0, s0, false, false);
      s1 = __builtin_amdgcn_wmma_f32_16x16x32_f16(false, qa[kd].v, false, b1.v, (short)0, s1, false, false);
    }

    // ---- online softmax update, P -> LDS (f16) ----
#pragma unroll
    for (int i = 0; i < 8; ++i) {
      float a0 = s0[i] * scale, a1 = s1[i] * scale;
      float v = fmaxf(a0, a1);
#pragma unroll
      for (int msk = 1; msk < 16; msk <<= 1) v = fmaxf(v, __shfl_xor(v, msk));
      float nm   = fmaxf(mx[i], v);
      float corr = __expf(mx[i] - nm);
      float p0 = __expf(a0 - nm), p1 = __expf(a1 - nm);
      float rs = p0 + p1;
#pragma unroll
      for (int msk = 1; msk < 16; msk <<= 1) rs += __shfl_xor(rs, msk);
      sm[i] = sm[i] * corr + rs;
      mx[i] = nm;
#pragma unroll
      for (int j = 0; j < 8; ++j) o[j][i] *= corr;
      int r = i + 8 * hs;
      pscr[wib][r * 32 + row16]      = (_Float16)p0;
      pscr[wib][r * 32 + 16 + row16] = (_Float16)p1;
    }

    asm volatile("s_wait_dscnt 0" ::: "memory");

    // reload P as A-fragment
    V16 pa;
    pa.p.lo = *(const v8h*)&pscr[wib][row16 * 32 + hs * 16];
    pa.p.hi = *(const v8h*)&pscr[wib][row16 * 32 + hs * 16 + 8];

    // ---- O += P @ V  (Vt is [dim][key] so B-fragments are contiguous) ----
#pragma unroll
    for (int j = 0; j < 8; ++j) {
      const half_t* vp = vbase + (size_t)(j * 16 + row16) * L + kt * 32 + hs * 16;
      V16 vb;
      vb.p.lo = *(const v8h*)vp;
      vb.p.hi = *(const v8h*)(vp + 8);
      o[j] = __builtin_amdgcn_wmma_f32_16x16x32_f16(false, pa.v, false, vb.v, (short)0, o[j], false, false);
    }
  }

  float* ob = out + ((size_t)b * L + qt * 16) * 512 + h * 128;
#pragma unroll
  for (int i = 0; i < 8; ++i) {
    int r = i + 8 * hs;
    float inv = 1.0f / sm[i];
#pragma unroll
    for (int j = 0; j < 8; ++j)
      ob[(size_t)r * 512 + j * 16 + row16] = o[j][i] * inv;
  }
}

// ---------------------------------------------------------------------------
// FSMN sliding mean: means[b,i,d] = mean(x[b, max(0,i-stride)..i, d])
// ---------------------------------------------------------------------------
__global__ void slide_mean_kernel(const float* __restrict__ x, float* __restrict__ out,
                                  int Bn, int L, int D, int stride)
{
  int t = blockIdx.x * blockDim.x + threadIdx.x;
  if (t >= Bn * D) return;
  int b = t / D, d = t % D;
  const float* xp = x + (size_t)b * L * D + d;
  float* op = out + (size_t)b * L * D + d;
  float sum = 0.0f;
  for (int i = 0; i < L; ++i) {
    sum += xp[(size_t)i * D];
    int drop = i - stride - 1;
    if (drop >= 0) sum -= xp[(size_t)drop * D];
    int cnt = (i < stride) ? (i + 1) : (stride + 1);
    op[(size_t)i * D] = sum / (float)cnt;
  }
}

// ---------------------------------------------------------------------------
// LayerNorm over D=512, one wave per row.
// ---------------------------------------------------------------------------
__global__ __launch_bounds__(256)
void ln_kernel(const float* __restrict__ x, const float* __restrict__ w,
               const float* __restrict__ bi, float* __restrict__ out, int M)
{
  int wave = (int)(blockIdx.x) * 8 + (threadIdx.x >> 5);
  int lane = threadIdx.x & 31;
  if (wave >= M) return;
  const float* xp = x + (size_t)wave * 512;
  float v[16];
  float s = 0.0f, s2 = 0.0f;
#pragma unroll
  for (int i = 0; i < 16; ++i) {
    v[i] = xp[lane + 32 * i];
    s += v[i]; s2 += v[i] * v[i];
  }
#pragma unroll
  for (int msk = 1; msk < 32; msk <<= 1) {
    s  += __shfl_xor(s, msk);
    s2 += __shfl_xor(s2, msk);
  }
  float mean = s * (1.0f / 512.0f);
  float var  = s2 * (1.0f / 512.0f) - mean * mean;
  float rstd = rsqrtf(var + 1e-5f);
  float* op = out + (size_t)wave * 512;
#pragma unroll
  for (int i = 0; i < 16; ++i) {
    int idx = lane + 32 * i;
    op[idx] = (v[i] - mean) * rstd * w[idx] + bi[idx];
  }
}

// ci[b,l,s] = h[b,l,:] . sf[b,s,:]
__global__ void ci_kernel(const float* __restrict__ h, const float* __restrict__ sf,
                          float* __restrict__ ci, int Bn, int L, int S)
{
  int t = blockIdx.x * blockDim.x + threadIdx.x;
  if (t >= Bn * L) return;
  int b = t / L;
  const float4* hp = (const float4*)(h + (size_t)t * 512);
  for (int s = 0; s < S; ++s) {
    const float4* sp = (const float4*)(sf + ((size_t)b * S + s) * 512);
    float acc = 0.0f;
    for (int k = 0; k < 128; ++k) {
      float4 a = hp[k], w = sp[k];
      acc += a.x * w.x + a.y * w.y + a.z * w.z + a.w * w.w;
    }
    ci[(size_t)t * S + s] = acc;
  }
}

// comb[row,n] += sum_s ci[row,s] * aw_f32[n*516 + s]
__global__ void adapter_ci_kernel(const float* __restrict__ ci, const float* __restrict__ aw,
                                  float* __restrict__ comb, int Mtot)
{
  int t = blockIdx.x * blockDim.x + threadIdx.x;
  if (t >= Mtot * 512) return;
  int row = t >> 9, n = t & 511;
  const float* c = ci + (size_t)row * 4;
  const float* w = aw + (size_t)n * 516;
  comb[t] += c[0] * w[0] + c[1] * w[1] + c[2] * w[2] + c[3] * w[3];
}

// qkv[B,L,3*512] -> qf/kf [b,h,l,128] f16 and vt [b,h,128,L] f16
__global__ void attn_prep_kernel(const float* __restrict__ qkv,
                                 half_t* __restrict__ qf, half_t* __restrict__ kf,
                                 half_t* __restrict__ vt, int Bn, int H, int L)
{
  size_t t = (size_t)blockIdx.x * blockDim.x + threadIdx.x;
  size_t total = (size_t)Bn * H * L * 128;
  if (t >= total) return;
  int d = (int)(t & 127);
  size_t r = t >> 7;
  int l = (int)(r % L);
  size_t bh = r / L;
  int h = (int)(bh % H);
  int b = (int)(bh / H);
  const float* src = qkv + ((size_t)b * L + l) * 1536 + h * 128 + d;
  qf[t] = (_Float16)src[0];
  kf[t] = (_Float16)src[512];
  vt[(bh * 128 + d) * L + l] = (_Float16)src[1024];
}

__global__ void cvt_f16_kernel(const float* __restrict__ s, half_t* __restrict__ d, long n)
{
  long i = (long)blockIdx.x * blockDim.x + threadIdx.x;
  if (i < n) d[i] = (_Float16)s[i];
}

// strided convert (for adapter: keep cols [soff, soff+cols) of each row)
__global__ void cvt_f16_strided_kernel(const float* __restrict__ s, half_t* __restrict__ d,
                                       int rows, int cols, int sld, int soff)
{
  int i = blockIdx.x * blockDim.x + threadIdx.x;
  if (i >= rows * cols) return;
  int r = i / cols, c = i % cols;
  d[i] = (_Float16)s[(size_t)r * sld + soff + c];
}

// ---------------------------------------------------------------------------
extern "C" void kernel_launch(void* const* d_in, const int* in_sizes, int n_in,
                              void* d_out, int out_size, void* d_ws, size_t ws_size,
                              hipStream_t stream)
{
  (void)in_sizes; (void)n_in; (void)out_size; (void)ws_size;
  const int B = 8, L = 2048, Fdim = 80, S = 4, H = 4;
  const int M = B * L;
  static const int ENC_STRIDES[8]  = {1, 2, 4, 8, 16, 32, 64, 128};
  const int POST_STRIDES[6] = {1, 2, 4, 8, 16, 32};

  const float* x        = (const float*)d_in[0];
  const float* spk_emb  = (const float*)d_in[1];
  const float* enc0_w   = (const float*)d_in[2];
  const float* enc0_b   = (const float*)d_in[3];
  const float* enc_lw   = (const float*)d_in[4];
  const float* enc_lb   = (const float*)d_in[5];
  const float* enc_mw   = (const float*)d_in[6];
  const float* enc_mb   = (const float*)d_in[7];
  const float* spk_w1   = (const float*)d_in[8];
  const float* spk_b1   = (const float*)d_in[9];
  const float* spk_w2   = (const float*)d_in[10];
  const float* spk_b2   = (const float*)d_in[11];
  const float* spk_w3   = (const float*)d_in[12];
  const float* spk_b3   = (const float*)d_in[13];
  const float* tr_in_w  = (const float*)d_in[14];
  const float* tr_in_b  = (const float*)d_in[15];
  const float* tr_out_w = (const float*)d_in[16];
  const float* tr_out_b = (const float*)d_in[17];
  const float* tr_ln1_w = (const float*)d_in[18];
  const float* tr_ln1_b = (const float*)d_in[19];
  const float* tr_ff1_w = (const float*)d_in[20];
  const float* tr_ff1_b = (const float*)d_in[21];
  const float* tr_ff2_w = (const float*)d_in[22];
  const float* tr_ff2_b = (const float*)d_in[23];
  const float* tr_ln2_w = (const float*)d_in[24];
  const float* tr_ln2_b = (const float*)d_in[25];
  const float* adp_w    = (const float*)d_in[26];
  const float* adp_b    = (const float*)d_in[27];
  const float* post_lw  = (const float*)d_in[28];
  const float* post_lb  = (const float*)d_in[29];
  const float* post_mw  = (const float*)d_in[30];
  const float* post_mb  = (const float*)d_in[31];
  const float* cls_w1   = (const float*)d_in[32];
  const float* cls_b1   = (const float*)d_in[33];
  const float* cls_w2   = (const float*)d_in[34];
  const float* cls_b2   = (const float*)d_in[35];
  float* outp = (float*)d_out;

  char* base = (char*)d_ws;
  size_t off = 0;
  auto carve = [&](size_t bytes) -> char* {
    char* p = base + off;
    off += (bytes + 255) & ~(size_t)255;
    return p;
  };

  // f16 weight arena
  half_t* wf_enc0  = (half_t*)carve((size_t)512 * 80 * 2);
  half_t* wf_elin  = (half_t*)carve((size_t)8 * 512 * 512 * 2);
  half_t* wf_emem  = (half_t*)carve((size_t)8 * 512 * 512 * 2);
  half_t* wf_spk1  = (half_t*)carve((size_t)512 * 192 * 2);
  half_t* wf_spk2  = (half_t*)carve((size_t)512 * 512 * 2);
  half_t* wf_spk3  = (half_t*)carve((size_t)512 * 512 * 2);
  half_t* wf_trin  = (half_t*)carve((size_t)4 * 1536 * 512 * 2);
  half_t* wf_trout = (half_t*)carve((size_t)4 * 512 * 512 * 2);
  half_t* wf_ff1   = (half_t*)carve((size_t)4 * 2048 * 512 * 2);
  half_t* wf_ff2   = (half_t*)carve((size_t)4 * 512 * 2048 * 2);
  half_t* wf_adp   = (half_t*)carve((size_t)512 * 512 * 2);
  half_t* wf_plin  = (half_t*)carve((size_t)6 * 512 * 512 * 2);
  half_t* wf_pmem  = (half_t*)carve((size_t)6 * 512 * 512 * 2);
  half_t* wf_cls1  = (half_t*)carve((size_t)512 * 512 * 2);
  half_t* wf_cls2  = (half_t*)carve((size_t)16 * 512 * 2);

  // f32 activation buffers
  float* BUF_H    = (float*)carve((size_t)M * 512 * 4);
  float* BUF_T    = (float*)carve((size_t)M * 512 * 4);
  float* BUF_MEAN = (float*)carve((size_t)M * 512 * 4);
  float* BUF_QKV  = (float*)carve((size_t)M * 1536 * 4);
  float* BUF_FF   = (float*)carve((size_t)M * 2048 * 4);
  float* CI       = (float*)carve((size_t)M * 4 * 4);
  float* SPK1     = (float*)carve((size_t)32 * 512 * 4);
  float* SPK2     = (float*)carve((size_t)32 * 512 * 4);
  float* SF       = (float*)carve((size_t)32 * 512 * 4);

  // qf/kf/vt live inside BUF_FF (free during attention)
  const size_t BHLHD = (size_t)B * H * L * 128;  // 8,388,608
  half_t* QF = (half_t*)BUF_FF;
  half_t* KF = QF + BHLHD;
  half_t* VT = KF + BHLHD;

  auto cvt = [&](const float* s, half_t* d, size_t n) {
    int blocks = (int)((n + 255) / 256);
    cvt_f16_kernel<<<blocks, 256, 0, stream>>>(s, d, (long)n);
  };
  auto gemm = [&](const float* A, int lda, const half_t* W, int ldw, const float* bias,
                  const float* res, int ldr, float* Out, int ldo,
                  int Mi, int Ni, int Ki, int relu) {
    int tilesM = (Mi + 31) / 32, tilesN = (Ni + 63) / 64;
    int waves  = tilesM * tilesN;
    int blocks = (waves + 7) / 8;
    gemm_ws_kernel<<<blocks, 256, 0, stream>>>(A, lda, W, ldw, bias, res, ldr, Out, ldo,
                                               Mi, Ni, Ki, relu);
  };

  // ---- convert weights to f16 ----
  cvt(enc0_w,  wf_enc0,  (size_t)512 * 80);
  cvt(enc_lw,  wf_elin,  (size_t)8 * 512 * 512);
  cvt(enc_mw,  wf_emem,  (size_t)8 * 512 * 512);
  cvt(spk_w1,  wf_spk1,  (size_t)512 * 192);
  cvt(spk_w2,  wf_spk2,  (size_t)512 * 512);
  cvt(spk_w3,  wf_spk3,  (size_t)512 * 512);
  cvt(tr_in_w, wf_trin,  (size_t)4 * 1536 * 512);
  cvt(tr_out_w,wf_trout, (size_t)4 * 512 * 512);
  cvt(tr_ff1_w,wf_ff1,   (size_t)4 * 2048 * 512);
  cvt(tr_ff2_w,wf_ff2,   (size_t)4 * 512 * 2048);
  cvt(post_lw, wf_plin,  (size_t)6 * 512 * 512);
  cvt(post_mw, wf_pmem,  (size_t)6 * 512 * 512);
  cvt(cls_w1,  wf_cls1,  (size_t)512 * 512);
  cvt(cls_w2,  wf_cls2,  (size_t)16 * 512);
  cvt_f16_strided_kernel<<<(512 * 512 + 255) / 256, 256, 0, stream>>>(
      adp_w, wf_adp, 512, 512, 516, 4);

  // ---- speech encoder: input proj + 8 FSMN layers ----
  gemm(x, Fdim, wf_enc0, Fdim, enc0_b, nullptr, 0, BUF_H, 512, M, 512, Fdim, 0);
  for (int i = 0; i < 8; ++i) {
    gemm(BUF_H, 512, wf_elin + (size_t)i * 512 * 512, 512, enc_lb + (size_t)i * 512,
         nullptr, 0, BUF_T, 512, M, 512, 512, 0);
    slide_mean_kernel<<<(B * 512 + 255) / 256, 256, 0, stream>>>(
        BUF_H, BUF_MEAN, B, L, 512, ENC_STRIDES[i]);
    gemm(BUF_MEAN, 512, wf_emem + (size_t)i * 512 * 512, 512, enc_mb + (size_t)i * 512,
         BUF_T, 512, BUF_H, 512, M, 512, 512, 0);
  }

  // ---- speaker MLP (M=32) ----
  gemm(spk_emb, 192, wf_spk1, 192, spk_b1, nullptr, 0, SPK1, 512, 32, 512, 192, 1);
  gemm(SPK1, 512, wf_spk2, 512, spk_b2, nullptr, 0, SPK2, 512, 32, 512, 512, 1);
  gemm(SPK2, 512, wf_spk3, 512, spk_b3, nullptr, 0, SF,   512, 32, 512, 512, 0);

  // ---- context-independent scores ----
  ci_kernel<<<(M + 255) / 256, 256, 0, stream>>>(BUF_H, SF, CI, B, L, S);

  // ---- 4-layer post-norm transformer (cd lives in BUF_H) ----
  for (int t = 0; t < 4; ++t) {
    gemm(BUF_H, 512, wf_trin + (size_t)t * 1536 * 512, 512, tr_in_b + (size_t)t * 1536,
         nullptr, 0, BUF_QKV, 1536, M, 1536, 512, 0);
    attn_prep_kernel<<<(int)(BHLHD / 256), 256, 0, stream>>>(BUF_QKV, QF, KF, VT, B, H, L);
    {
      int waves = B * H * (L / 16);
      attn_kernel<<<(waves + 7) / 8, 256, 0, stream>>>(QF, KF, VT, BUF_T, B, H, L);
    }
    gemm(BUF_T, 512, wf_trout + (size_t)t * 512 * 512, 512, tr_out_b + (size_t)t * 512,
         BUF_H, 512, BUF_QKV, 512, M, 512, 512, 0);
    ln_kernel<<<(M + 7) / 8, 256, 0, stream>>>(BUF_QKV, tr_ln1_w + (size_t)t * 512,
                                               tr_ln1_b + (size_t)t * 512, BUF_H, M);
    gemm(BUF_H, 512, wf_ff1 + (size_t)t * 2048 * 512, 512, tr_ff1_b + (size_t)t * 2048,
         nullptr, 0, BUF_FF, 2048, M, 2048, 512, 1);
    gemm(BUF_FF, 2048, wf_ff2 + (size_t)t * 512 * 2048, 2048, tr_ff2_b + (size_t)t * 512,
         BUF_H, 512, BUF_QKV, 512, M, 512, 2048, 0);
    ln_kernel<<<(M + 7) / 8, 256, 0, stream>>>(BUF_QKV, tr_ln2_w + (size_t)t * 512,
                                               tr_ln2_b + (size_t)t * 512, BUF_H, M);
  }

  // ---- adapter: comb = [ci, cd] @ adapter_w^T + b ----
  gemm(BUF_H, 512, wf_adp, 512, adp_b, nullptr, 0, BUF_T, 512, M, 512, 512, 0);
  adapter_ci_kernel<<<(int)(((size_t)M * 512) / 256), 256, 0, stream>>>(CI, adp_w, BUF_T, M);

  // ---- 6 post FSMN layers (comb in BUF_T) ----
  for (int i = 0; i < 6; ++i) {
    gemm(BUF_T, 512, wf_plin + (size_t)i * 512 * 512, 512, post_lb + (size_t)i * 512,
         nullptr, 0, BUF_H, 512, M, 512, 512, 0);
    slide_mean_kernel<<<(B * 512 + 255) / 256, 256, 0, stream>>>(
        BUF_T, BUF_MEAN, B, L, 512, POST_STRIDES[i]);
    gemm(BUF_MEAN, 512, wf_pmem + (size_t)i * 512 * 512, 512, post_mb + (size_t)i * 512,
         BUF_H, 512, BUF_T, 512, M, 512, 512, 0);
  }

  // ---- classifier ----
  gemm(BUF_T, 512, wf_cls1, 512, cls_b1, nullptr, 0, BUF_H, 512, M, 512, 512, 1);
  gemm(BUF_H, 512, wf_cls2, 512, cls_b2, nullptr, 0, outp, 16, M, 16, 512, 0);
}
